// MGCNEncoderOfficial_17626545783584
// MI455X (gfx1250) — matrix operand order repaired and checked by reading, not compiled, not run on hardware
//
#include <hip/hip_runtime.h>
#include <hip/hip_bf16.h>
#include <stdint.h>

// ---------------------------------------------------------------------------
// MGCN encoder forward for MI455X (gfx1250, wave32, WMMA).
// - All dense layers: v_wmma_f32_16x16x32_bf16, f32 accumulate.
// - Weights pre-packed once per launch into bf16 K-pair dwords (zero-padded to
//   N multiple of 64) -> GEMM has ZERO bounds checks and the B tile is fetched
//   with global_load_async_to_lds_b128 (double-buffered, s_wait_asynccnt).
// - f32->bf16 pair pack: 2x v_add + 1x v_perm_b32 (cvt_pk_bf16 builtin absent).
// - The (B,C,T,FT) fused tensor (134 MB) is never materialized.
// ---------------------------------------------------------------------------

#define Bdim 128
#define Cdim 4
#define Tdim 1024
#define DM   512
#define DFF  2048
#define DI   512
#define DS   32
#define DR   32
#define KC   2
#define FCdim 64
#define FTdim 64
#define ROWS (Bdim*Cdim)     // 512 activation rows
#define DBLLD 128            // dbl buffer row stride (96 padded to 128)

typedef __bf16 bf16_t;
typedef bf16_t v16bf __attribute__((ext_vector_type(16)));
typedef float  v8f   __attribute__((ext_vector_type(8)));

// pack two f32 into (bf16(a) | bf16(b)<<16): round-half-up + one v_perm_b32
__device__ __forceinline__ unsigned int pk2(float a, float b) {
  const unsigned ua = __builtin_bit_cast(unsigned int, a) + 0x8000u;
  const unsigned ub = __builtin_bit_cast(unsigned int, b) + 0x8000u;
  // {s0=ub -> bytes 4..7, s1=ua -> bytes 0..3}; dest = ua[3:2] , ub[3:2]
  return __builtin_amdgcn_perm(ub, ua, 0x07060302u);
}

// ACT: 0 none, 1 relu, 2 gelu(exact), 3 silu, 4 softplus
template<int ACT> __device__ __forceinline__ float act_fn(float x) {
  if (ACT == 1) return x > 0.f ? x : 0.f;
  if (ACT == 2) return 0.5f * x * (1.f + erff(x * 0.70710678118654752f));
  if (ACT == 3) return x / (1.f + __expf(-x));
  if (ACT == 4) return (x > 20.f) ? x : log1pf(__expf(x));
  return x;
}

// ---------------------------------------------------------------------------
// Weight pre-pack: W (KxN f32, row-major) -> Wp (K/2 x Np dwords),
// Wp[kp*Np+n] = (bf16(W[2kp][n]), bf16(W[2kp+1][n])), zero-padded for n>=N.
// This is exactly the WMMA B-fragment pair layout, so GEMM B tiles are a raw
// byte copy (async-to-LDS capable).
// ---------------------------------------------------------------------------
__global__ void pack_weight_kernel(const float* __restrict__ W,
                                   unsigned int* __restrict__ Wp,
                                   int K, int N, int Np) {
  const int idx = blockIdx.x * blockDim.x + threadIdx.x;
  const int total = (K >> 1) * Np;
  if (idx >= total) return;
  const int kp = idx / Np, n = idx - kp * Np;
  float v0 = 0.f, v1 = 0.f;
  if (n < N) {
    v0 = W[(size_t)(2 * kp) * N + n];
    v1 = W[(size_t)(2 * kp + 1) * N + n];
  }
  Wp[idx] = pk2(v0, v1);
}

// ---------------------------------------------------------------------------
// WMMA bf16 GEMM (tile-exact, no guards): C = act(A @ B + bias [+ C if ACC])
// A: MxK f32 (lda). Bp: packed bf16 pairs, K/2 rows x N dwords. C: MxN (ldc).
// Requires: M%64==0, N%64==0, K%32==0 (guaranteed by pre-padding).
// 256 threads = 8 waves; block tile 64x64; wave strip 16x32 (2 accumulators).
// B tile: 256 x global_load_async_to_lds_b128, double-buffered across K.
// ---------------------------------------------------------------------------
#define BM 64
#define BN 64
#define BK 32
#define ALD 34        // bf16 elems per A row in LDS (+2 pad)

template<int ACT, bool ACC>
__global__ __launch_bounds__(256) void gemm_wmma_kernel(
    const float* __restrict__ A, const unsigned int* __restrict__ Bp,
    const float* __restrict__ bias, float* __restrict__ C,
    int M, int N, int K, int lda, int ldc)
{
  (void)M;
  __shared__ unsigned short Alds[BM * ALD];
  __shared__ __align__(16) unsigned int Blds[2][16 * 64];  // XOR-swizzled

  const int tid   = threadIdx.x;
  const int lane  = tid & 31;
  const int w     = tid >> 5;
  const int m0    = blockIdx.y * BM;
  const int n0    = blockIdx.x * BN;
  const int wm    = (w >> 1) * 16;
  const int wn    = (w & 1) * 32;
  const int ncol  = lane & 15;
  const int khalf = lane >> 4;

  // one b128 async copy per thread per K-tile; row = local kp, ch = dword col
  const int brow = tid >> 4;
  const int bch  = (tid & 15) << 2;
  const int bswz = (brow >> 3) << 5;           // XOR bit5 on kp-bit3
  auto issue_b = [&](int kk0, unsigned int* blbuf) {
    const unsigned voff = (unsigned)((((size_t)((kk0 >> 1) + brow) * N) + n0 + bch) * 4u);
    unsigned int* dstp  = blbuf + brow * 64 + (bch ^ bswz);
    const unsigned dst  = (unsigned)(uintptr_t)dstp;    // low 32b = LDS offset
    asm volatile("global_load_async_to_lds_b128 %0, %1, %2"
                 :: "v"(dst), "v"(voff), "s"(Bp) : "memory");
  };

  // ---- accumulator init ----
  v8f c0, c1;
  {
    const int col0  = n0 + wn + ncol;
    const int col1  = col0 + 16;
    const int rbase = m0 + wm + (khalf << 3);
    float b0 = 0.f, b1 = 0.f;
    if (bias) { b0 = bias[col0]; b1 = bias[col1]; }
#pragma unroll
    for (int v = 0; v < 8; ++v) {
      float a0 = b0, a1 = b1;
      if (ACC) {
        a0 += C[(size_t)(rbase + v) * ldc + col0];
        a1 += C[(size_t)(rbase + v) * ldc + col1];
      }
      c0[v] = a0; c1[v] = a1;
    }
  }

  issue_b(0, Blds[0]);                         // prime the pipeline
  int buf = 0;

  for (int k0 = 0; k0 < K; k0 += BK) {
    // ---- A tile 64x32: float4 loads -> pk2 -> LDS (2 per thread) ----
#pragma unroll
    for (int i = 0; i < 2; ++i) {
      const int idx = tid + 256 * i;           // 0..511
      const int mm = idx >> 3, kk = (idx & 7) << 2;
      const float4 va = *(const float4*)(A + (size_t)(m0 + mm) * lda + k0 + kk);
      unsigned int* dp = (unsigned int*)&Alds[mm * ALD + kk];
      dp[0] = pk2(va.x, va.y);
      dp[1] = pk2(va.z, va.w);
    }
    const bool more = (k0 + BK < K);
    if (more) issue_b(k0 + BK, Blds[buf ^ 1]); // overlap next B tile
    asm volatile("s_wait_xcnt 0x0" ::: "memory");
    if (more) asm volatile("s_wait_asynccnt 0x20" ::: "memory"); // cur tile done
    else      asm volatile("s_wait_asynccnt 0x0"  ::: "memory");
    __syncthreads();

    // ---- fragments (wave32, k = 16*khalf + s in both A and B) ----
    union { unsigned int u[8]; v16bf v; } af, bf0, bf1;
    {
      const unsigned short* ap = &Alds[(wm + ncol) * ALD];
#pragma unroll
      for (int j = 0; j < 8; ++j) {
        af.u[j]     = *(const unsigned int*)(ap + 16 * khalf + 2 * j);       // K 16*khalf + 0..15
      }
      const unsigned int* bl = Blds[buf];
      const int sw = khalf << 5;
#pragma unroll
      for (int j = 0; j < 8; ++j) {
        const int kp = 8 * khalf + j;
        bf0.u[j] = bl[kp * 64 + ((wn + ncol) ^ sw)];
        bf1.u[j] = bl[kp * 64 + ((wn + 16 + ncol) ^ sw)];
      }
    }
    c0 = __builtin_amdgcn_wmma_f32_16x16x32_bf16(false, af.v, false, bf0.v,
                                                 (short)0, c0, false, false);
    c1 = __builtin_amdgcn_wmma_f32_16x16x32_bf16(false, af.v, false, bf1.v,
                                                 (short)0, c1, false, false);
    __syncthreads();                            // Alds reused next iter
    buf ^= 1;
  }

  // ---- epilogue ----
  {
    const int col0  = n0 + wn + ncol;
    const int col1  = col0 + 16;
    const int rbase = m0 + wm + (khalf << 3);
#pragma unroll
    for (int v = 0; v < 8; ++v) {
      C[(size_t)(rbase + v) * ldc + col0] = act_fn<ACT>(c0[v]);
      C[(size_t)(rbase + v) * ldc + col1] = act_fn<ACT>(c1[v]);
    }
  }
}

template<int ACT, bool ACC>
static inline void gemm(const float* A, const unsigned int* Bp, const float* bias,
                        float* C, int M, int N, int K, int lda, int ldc,
                        hipStream_t s) {
  dim3 g(N / BN, M / BM);
  gemm_wmma_kernel<ACT, ACC><<<g, dim3(256), 0, s>>>(A, Bp, bias, C, M, N, K, lda, ldc);
}

// ---------------------------------------------------------------------------
// Tiny / elementwise kernels
// ---------------------------------------------------------------------------
__global__ void adj_norm_kernel(const float* __restrict__ adj, float* __restrict__ adjn) {
  if (threadIdx.x == 0 && blockIdx.x == 0) {
    float a[Cdim][Cdim], dis[Cdim];
    for (int i = 0; i < Cdim; ++i)
      for (int j = 0; j < Cdim; ++j) {
        float v = adj[i * Cdim + j]; v = v > 0.f ? v : 0.f;
        a[i][j] = v + (i == j ? 1.f : 0.f);
      }
    for (int i = 0; i < Cdim; ++i) {
      float s = 0.f;
      for (int j = 0; j < Cdim; ++j) s += a[i][j];
      dis[i] = rsqrtf(s + 1e-6f);
    }
    for (int i = 0; i < Cdim; ++i)
      for (int j = 0; j < Cdim; ++j)
        adjn[i * Cdim + j] = dis[i] * a[i][j] * dis[j];
  }
}

// Chebyshev supports stacked as S[row=(b*C+n)][k*T + t], k=0..2 (K=3072)
__global__ void supports_kernel(const float* __restrict__ x, const float* __restrict__ adjn,
                                float* __restrict__ S) {
  const int idx = blockIdx.x * blockDim.x + threadIdx.x;   // B*T
  if (idx >= Bdim * Tdim) return;
  const int b = idx >> 10, t = idx & (Tdim - 1);
  float t0[Cdim], t1[Cdim], t2[Cdim];
  for (int m = 0; m < Cdim; ++m) t0[m] = x[(size_t)(b * Cdim + m) * Tdim + t];
  for (int n = 0; n < Cdim; ++n) {
    float s = 0.f;
    for (int m = 0; m < Cdim; ++m) s += adjn[n * Cdim + m] * t0[m];
    t1[n] = s;
  }
  for (int n = 0; n < Cdim; ++n) {
    float s = 0.f;
    for (int m = 0; m < Cdim; ++m) s += adjn[n * Cdim + m] * t1[m];
    t2[n] = 2.f * s - t0[n];
  }
  for (int n = 0; n < Cdim; ++n) {
    float* row = S + (size_t)(b * Cdim + n) * (3 * Tdim);
    row[t] = t0[n]; row[Tdim + t] = t1[n]; row[2 * Tdim + t] = t2[n];
  }
}

// causal depthwise conv (KC=2) + SiLU over seq length L=C=4.
template<bool REV>
__global__ void conv_silu_kernel(const float* __restrict__ xz, const float* __restrict__ cw,
                                 const float* __restrict__ cb, float* __restrict__ xs) {
  const int idx = blockIdx.x * blockDim.x + threadIdx.x;   // B*DI
  if (idx >= Bdim * DI) return;
  const int b = idx >> 9, d = idx & (DI - 1);
  const float w0 = cw[d * KC + 0], w1 = cw[d * KC + 1], bb = cb[d];
  float prev = 0.f;
  for (int l = 0; l < Cdim; ++l) {
    const int sl = REV ? (Cdim - 1 - l) : l;
    const float cur = xz[(size_t)(b * Cdim + sl) * (2 * DI) + d];
    const float xc = bb + prev * w0 + cur * w1;
    xs[(size_t)(b * Cdim + l) * DI + d] = xc / (1.f + __expf(-xc));
    prev = cur;
  }
}

// selective-scan (L=4, DS=32 state in registers) + SiLU(z) gate, de-reversed.
template<bool REV>
__global__ void scan_kernel(const float* __restrict__ dbl, const float* __restrict__ dt,
                            const float* __restrict__ xs,  const float* __restrict__ xz,
                            const float* __restrict__ A_log, const float* __restrict__ Dp,
                            float* __restrict__ yout) {
  const int idx = blockIdx.x * blockDim.x + threadIdx.x;   // B*DI
  if (idx >= Bdim * DI) return;
  const int b = idx >> 9, d = idx & (DI - 1);
  float A[DS], h[DS];
#pragma unroll
  for (int s = 0; s < DS; ++s) { A[s] = -__expf(A_log[d * DS + s]); h[s] = 0.f; }
  const float Dd = Dp[d];
  for (int l = 0; l < Cdim; ++l) {
    const int r = b * Cdim + l;
    const float dtv = dt[(size_t)r * DI + d];
    const float xsv = xs[(size_t)r * DI + d];
    const float dx  = dtv * xsv;
    const float* Bm = dbl + (size_t)r * DBLLD + DR;
    const float* Cm = Bm + DS;
    float y = 0.f;
#pragma unroll
    for (int s = 0; s < DS; ++s) {
      h[s] = __expf(dtv * A[s]) * h[s] + dx * Bm[s];
      y += h[s] * Cm[s];
    }
    y += xsv * Dd;
    const int sl = REV ? (Cdim - 1 - l) : l;
    const float z = xz[(size_t)(b * Cdim + sl) * (2 * DI) + DI + d];
    y *= z / (1.f + __expf(-z));
    yout[(size_t)(b * Cdim + sl) * DI + d] = y;
  }
}

// LayerNorm over last dim D; optional residual add. dst may alias src.
__global__ __launch_bounds__(256) void ln_kernel(
    float* __restrict__ dst, const float* __restrict__ src, const float* __restrict__ res,
    const float* __restrict__ g, const float* __restrict__ bt, int D) {
  const int row = blockIdx.x;
  const float* sp = src + (size_t)row * D;
  const float* rp = res ? res + (size_t)row * D : nullptr;
  float lsum = 0.f, lsq = 0.f;
  for (int i = threadIdx.x; i < D; i += blockDim.x) {
    const float v = sp[i] + (rp ? rp[i] : 0.f);
    lsum += v; lsq += v * v;
  }
#pragma unroll
  for (int o = 16; o > 0; o >>= 1) {          // wave32 reduction
    lsum += __shfl_xor(lsum, o, 32);
    lsq  += __shfl_xor(lsq,  o, 32);
  }
  __shared__ float rs[8], rq[8];
  __shared__ float mean_s, inv_s;
  const int w = threadIdx.x >> 5, lane = threadIdx.x & 31;
  if (lane == 0) { rs[w] = lsum; rq[w] = lsq; }
  __syncthreads();
  if (threadIdx.x == 0) {
    float s = 0.f, q = 0.f;
    const int nw = blockDim.x >> 5;
    for (int i = 0; i < nw; ++i) { s += rs[i]; q += rq[i]; }
    const float m = s / D;
    mean_s = m;
    inv_s  = rsqrtf(q / D - m * m + 1e-5f);
  }
  __syncthreads();
  const float m = mean_s, inv = inv_s;
  float* dp = dst + (size_t)row * D;
  for (int i = threadIdx.x; i < D; i += blockDim.x) {
    const float v = sp[i] + (rp ? rp[i] : 0.f);
    dp[i] = (v - m) * inv * g[i] + bt[i];
  }
}

// Fused: up-proj + residual-conv + ReLU + LayerNorm(FT) + pd_W dot, in regs.
__global__ __launch_bounds__(256) void fuse_kernel(
    const float* __restrict__ x, const float* __restrict__ mo,
    const float* __restrict__ puW, const float* __restrict__ puB,
    const float* __restrict__ rcw, const float* __restrict__ rcb,
    const float* __restrict__ fng, const float* __restrict__ fnb,
    const float* __restrict__ pdW, const float* __restrict__ pdB,
    float* __restrict__ down) {
  __shared__ float s_rcw[FTdim], s_rcb[FTdim], s_puW[FTdim], s_puB[FTdim];
  __shared__ float s_fng[FTdim], s_fnb[FTdim], s_pdW[FTdim];
  if (threadIdx.x < FTdim) {
    const int j = threadIdx.x;
    s_rcw[j] = rcw[j]; s_rcb[j] = rcb[j]; s_puW[j] = puW[j]; s_puB[j] = puB[j];
    s_fng[j] = fng[j]; s_fnb[j] = fnb[j]; s_pdW[j] = pdW[j];
  }
  __syncthreads();
  const int idx = blockIdx.x * blockDim.x + threadIdx.x;   // B*C*T
  if (idx >= Bdim * Cdim * Tdim) return;
  const float xv = x[idx], mv = mo[idx];
  float f[FTdim], s = 0.f, q = 0.f;
#pragma unroll
  for (int j = 0; j < FTdim; ++j) {
    float v = xv * s_rcw[j] + s_rcb[j] + mv * s_puW[j] + s_puB[j];
    v = v > 0.f ? v : 0.f;
    f[j] = v; s += v; q += v * v;
  }
  const float m   = s * (1.f / FTdim);
  const float inv = rsqrtf(q * (1.f / FTdim) - m * m + 1e-5f);
  float acc = 0.f;
#pragma unroll
  for (int j = 0; j < FTdim; ++j)
    acc += ((f[j] - m) * inv * s_fng[j] + s_fnb[j]) * s_pdW[j];
  down[idx] = acc + pdB[0];
}

__global__ void pool_kernel(const float* __restrict__ down, float* __restrict__ pooled) {
  const int idx = blockIdx.x * blockDim.x + threadIdx.x;   // B*T
  if (idx >= Bdim * Tdim) return;
  const int b = idx >> 10, t = idx & (Tdim - 1);
  float s = 0.f;
  for (int c = 0; c < Cdim; ++c) s += down[(size_t)(b * Cdim + c) * Tdim + t];
  pooled[idx] = s * (1.f / Cdim);
}

// ---------------------------------------------------------------------------
// Orchestration
// ---------------------------------------------------------------------------
extern "C" void kernel_launch(void* const* d_in, const int* in_sizes, int n_in,
                              void* d_out, int out_size, void* d_ws, size_t ws_size,
                              hipStream_t stream) {
  (void)in_sizes; (void)n_in; (void)out_size; (void)ws_size;
  auto F = [&](int i) { return (const float*)d_in[i]; };

  // Flattening: recursive insertion-order traversal of setup_inputs() dict.
  const float* x      = F(0);
  const float* p_adj  = F(1);
  const float* chebW  = F(2);   const float* chebB = F(3);
  const float* stgW   = F(4);   const float* stgB  = F(5);
  const float* embW   = F(6);   const float* embB  = F(7);
  const int LBASE = 8, LSTRIDE = 26, MSTRIDE = 9;
  const float* enormG = F(60);  const float* enormB = F(61);
  const float* ptW    = F(62);  const float* ptB    = F(63);
  const float* puW    = F(64);  const float* puB    = F(65);
  const float* rcw    = F(66);  const float* rcb    = F(67);
  const float* fng    = F(68);  const float* fnb    = F(69);
  const float* pdW    = F(70);  const float* pdB    = F(71);
  const float* outW   = F(72);  const float* outB   = F(73);

  // ---- workspace: f32 activations then packed bf16 weights (~43 MB total)
  float* ws = (float*)d_ws;
  size_t off = 0;
  auto alloc = [&](size_t n) { float* p = ws + off; off += n; return p; };
  float* adjn  = alloc(Cdim * Cdim);
  float* S     = alloc((size_t)ROWS * 3 * Tdim);
  float* chb   = alloc((size_t)ROWS * FCdim);
  float* gcn   = alloc((size_t)ROWS * Tdim);
  float* enc   = alloc((size_t)ROWS * DM);
  float* xzb   = alloc((size_t)ROWS * 2 * DI);
  float* xsb   = alloc((size_t)ROWS * DI);
  float* dblb  = alloc((size_t)ROWS * DBLLD);
  float* dtb_  = alloc((size_t)ROWS * DI);
  float* yb    = alloc((size_t)ROWS * DI);
  float* newb  = alloc((size_t)ROWS * DM);
  float* ffn   = alloc((size_t)ROWS * DFF);
  float* mo    = alloc((size_t)ROWS * Tdim);
  float* downb = alloc((size_t)ROWS * Tdim);
  float* poolb = alloc((size_t)Bdim * Tdim);

  unsigned int* wsu = (unsigned int*)(ws + off);
  size_t uoff = 0;
  auto pack = [&](const float* W, int K, int N, int Np) {
    unsigned int* dst = wsu + uoff;
    uoff += (size_t)(K >> 1) * Np;
    const int total = (K >> 1) * Np;
    pack_weight_kernel<<<(total + 255) / 256, 256, 0, stream>>>(W, dst, K, N, Np);
    return (const unsigned int*)dst;
  };

  // pack every weight once (bf16, N padded to multiple of 64)
  const unsigned int* chebWp = pack(chebW, 3 * Tdim, FCdim, FCdim);
  const unsigned int* stgWp  = pack(stgW,  FCdim, Tdim, Tdim);
  const unsigned int* embWp  = pack(embW,  Tdim, DM, DM);
  const unsigned int* ptWp   = pack(ptW,   DM, Tdim, Tdim);
  const unsigned int* outWp  = pack(outW,  Tdim, Bdim, Bdim);
  const unsigned int* ipP[2][2];   // [layer][dir]
  const unsigned int* xpP[2][2];
  const unsigned int* dtP[2][2];
  const unsigned int* opP[2][2];
  const unsigned int* c1P[2];
  const unsigned int* c2P[2];
  for (int L = 0; L < 2; ++L) {
    const int lb = LBASE + L * LSTRIDE;
    for (int dir = 0; dir < 2; ++dir) {
      const int mb = lb + dir * MSTRIDE;
      ipP[L][dir] = pack(F(mb + 0), DM, 2 * DI, 2 * DI);
      xpP[L][dir] = pack(F(mb + 3), DI, DR + 2 * DS, DBLLD);   // 96 -> 128 pad
      dtP[L][dir] = pack(F(mb + 4), DR, DI, DI);
      opP[L][dir] = pack(F(mb + 8), DI, DM, DM);
    }
    c1P[L] = pack(F(lb + 18), DM, DFF, DFF);
    c2P[L] = pack(F(lb + 20), DFF, DM, DM);
  }

  const dim3 B256(256);
  // --- graph frontend ---
  adj_norm_kernel<<<1, 1, 0, stream>>>(p_adj, adjn);
  supports_kernel<<<(Bdim * Tdim + 255) / 256, B256, 0, stream>>>(x, adjn, S);
  gemm<1, false>(S, chebWp, chebB, chb, ROWS, FCdim, 3 * Tdim, 3 * Tdim, FCdim, stream);
  gemm<0, false>(chb, stgWp, stgB, gcn, ROWS, Tdim, FCdim, FCdim, Tdim, stream);
  gemm<0, false>(gcn, embWp, embB, enc, ROWS, DM, Tdim, Tdim, DM, stream);

  // --- BiMamba layers ---
  for (int L = 0; L < 2; ++L) {
    const int lb = LBASE + L * LSTRIDE;
    for (int dir = 0; dir < 2; ++dir) {
      const int mb = lb + dir * MSTRIDE;
      const float* cw   = F(mb + 1);
      const float* cb   = F(mb + 2);
      const float* dtbv = F(mb + 5);
      const float* Alog = F(mb + 6);
      const float* Dp   = F(mb + 7);

      gemm<0, false>(enc, ipP[L][dir], nullptr, xzb, ROWS, 2 * DI, DM, DM, 2 * DI, stream);
      if (dir == 0) conv_silu_kernel<false><<<(Bdim * DI + 255) / 256, B256, 0, stream>>>(xzb, cw, cb, xsb);
      else          conv_silu_kernel<true ><<<(Bdim * DI + 255) / 256, B256, 0, stream>>>(xzb, cw, cb, xsb);
      gemm<0, false>(xsb, xpP[L][dir], nullptr, dblb, ROWS, DBLLD, DI, DI, DBLLD, stream);
      gemm<4, false>(dblb, dtP[L][dir], dtbv, dtb_, ROWS, DI, DR, DBLLD, DI, stream);
      if (dir == 0) scan_kernel<false><<<(Bdim * DI + 255) / 256, B256, 0, stream>>>(dblb, dtb_, xsb, xzb, Alog, Dp, yb);
      else          scan_kernel<true ><<<(Bdim * DI + 255) / 256, B256, 0, stream>>>(dblb, dtb_, xsb, xzb, Alog, Dp, yb);
      if (dir == 0) gemm<0, false>(yb, opP[L][dir], nullptr, newb, ROWS, DM, DI, DI, DM, stream);
      else          gemm<0, true >(yb, opP[L][dir], nullptr, newb, ROWS, DM, DI, DI, DM, stream);
    }
    ln_kernel<<<ROWS, B256, 0, stream>>>(enc, enc, newb, F(lb + 22), F(lb + 23), DM);
    gemm<2, false>(enc, c1P[L], F(lb + 19), ffn, ROWS, DFF, DM, DM, DFF, stream);
    gemm<0, false>(ffn, c2P[L], F(lb + 21), newb, ROWS, DM, DFF, DFF, DM, stream);
    ln_kernel<<<ROWS, B256, 0, stream>>>(enc, enc, newb, F(lb + 24), F(lb + 25), DM);
  }

  // --- head ---
  ln_kernel<<<ROWS, B256, 0, stream>>>(enc, enc, nullptr, enormG, enormB, DM);
  gemm<0, false>(enc, ptWp, ptB, mo, ROWS, Tdim, DM, DM, Tdim, stream);
  fuse_kernel<<<(Bdim * Cdim * Tdim + 255) / 256, B256, 0, stream>>>(
      x, mo, puW, puB, rcw, rcb, fng, fnb, pdW, pdB, downb);
  pool_kernel<<<(Bdim * Tdim + 255) / 256, B256, 0, stream>>>(downb, poolb);
  gemm<0, false>(poolb, outWp, outB, (float*)d_out, Bdim, Bdim, Tdim, Tdim, Bdim, stream);
}
